// GCN_68152541053088
// MI455X (gfx1250) — matrix operand order repaired and checked by reading, not compiled, not run on hardware
//
#include <hip/hip_runtime.h>
#include <hip/hip_bf16.h>
#include <math.h>

// Problem constants (from reference)
#define BS   32
#define OBJ  128
#define D    256
#define RR   32
#define EE   64
#define DEPTH 3
#define KCAT (OBJ + RR)      // 160 : [Asum_obj | Asum_rel] K dimension
#define NCAT (2 * D)         // 512 : [W1 | W2a] N dimension

typedef __attribute__((ext_vector_type(16))) _Float16 v16h;
typedef __attribute__((ext_vector_type(8)))  _Float16 v8h;
typedef __attribute__((ext_vector_type(8)))  float    v8f;

// ---------------------------------------------------------------------------
// WMMA fragment loaders (wave32, V_WMMA_F32_16X16X32_F16)
// A (16x32, 16-bit, row-major, ld = lda):
//   lanes 0-15 : M = lane,     halfs 0-7 = K 0..7,  halfs 8-15 = K 16..23
//   lanes16-31 : M = lane-16,  halfs 0-7 = K 8..15, halfs 8-15 = K 24..31
// ---------------------------------------------------------------------------
__device__ __forceinline__ v16h load_a16(const _Float16* __restrict__ A, int lda,
                                         int m0, int k0) {
  const int lane = threadIdx.x & 31;
  const int m  = m0 + (lane & 15);
  const int kb = k0 + ((lane >> 4) << 3);
  const _Float16* p = A + m * lda + kb;
  v8h lo = *(const v8h*)(p);
  v8h hi = *(const v8h*)(p + 16);
  return __builtin_shufflevector(lo, hi, 0,1,2,3,4,5,6,7,8,9,10,11,12,13,14,15);
}

// B (32x16) loaded from K-contiguous (transposed) storage Bt[n][k], ld = ldk:
//   lanes 0-15 : N = lane,     halfs 0-7 = K 0..7,  halfs 8-15 = K 16..23
//   lanes16-31 : N = lane-16,  halfs 0-7 = K 8..15, halfs 8-15 = K 24..31
__device__ __forceinline__ v16h load_b16t(const _Float16* __restrict__ Bt, int ldk,
                                          int n0, int k0) {
  const int lane = threadIdx.x & 31;
  const int n  = n0 + (lane & 15);
  const int kb = k0 + ((lane >> 4) << 3);
  const _Float16* p = Bt + n * ldk + kb;
  v8h lo = *(const v8h*)(p);
  v8h hi = *(const v8h*)(p + 16);
  return __builtin_shufflevector(lo, hi, 0,1,2,3,4,5,6,7,8,9,10,11,12,13,14,15);
}

__device__ __forceinline__ v8f wmma_f16(v16h a, v16h b, v8f c) {
  return __builtin_amdgcn_wmma_f32_16x16x32_f16(false, a, false, b, (short)0, c,
                                                false, false);
}

// ---------------------------------------------------------------------------
// One-time: Asum_obj[b,o,o'] = sum_r A[b,o,o'*R+r]   -> AsumCat[b][o][o']
//           Asum_rel[b,o,r ] = sum_o' A[b,o,o'*R+r]  -> AsumCat[b][o][128+r]
// grid (OBJ, BS), block 128 (thread = o')
// ---------------------------------------------------------------------------
__global__ void k_prep_asum(const float* __restrict__ A, _Float16* __restrict__ Asum) {
  const int o = blockIdx.x, b = blockIdx.y, t = threadIdx.x;
  __shared__ float lds[OBJ][RR + 1];
  const float* row = A + ((size_t)(b * OBJ + o)) * (RR * OBJ) + t * RR;
  float vals[RR];
#pragma unroll
  for (int j = 0; j < RR / 4; ++j) {
    float4 v = ((const float4*)row)[j];
    vals[4 * j + 0] = v.x; vals[4 * j + 1] = v.y;
    vals[4 * j + 2] = v.z; vals[4 * j + 3] = v.w;
  }
  float s = 0.f;
#pragma unroll
  for (int r = 0; r < RR; ++r) { s += vals[r]; lds[t][r] = vals[r]; }
  _Float16* dst = Asum + ((size_t)(b * OBJ + o)) * KCAT;
  dst[t] = (_Float16)s;
  __syncthreads();
  if (t < RR) {
    float sr = 0.f;
#pragma unroll 8
    for (int o2 = 0; o2 < OBJ; ++o2) sr += lds[o2][t];
    dst[OBJ + t] = (_Float16)sr;
  }
}

// f32 -> f16 convert (x seed)
__global__ void k_cvt_f16(const float* __restrict__ in, _Float16* __restrict__ out, int n) {
  int i = blockIdx.x * blockDim.x + threadIdx.x;
  if (i < n) out[i] = (_Float16)in[i];
}

// Per depth: WcatT[n][k] (K-contiguous) = n<256 ? W1[k][n] : W2a[k][n-256]
// grid 512, block 256 (thread = k)
__global__ void k_wcvt(const float* __restrict__ W1, const float* __restrict__ W2,
                       _Float16* __restrict__ WcatT) {
  const int n = blockIdx.x, k = threadIdx.x;
  const float v = (n < D) ? W1[k * D + n] : W2[k * D + (n - D)];
  WcatT[n * D + k] = (_Float16)v;
}

// Per depth: relW[r][f] = b2[f] + sum_e rel[r][e] * W2b[e][f]; broadcast into
// every batch's H2t at k = 128+r (K-contiguous layout H2t[b][n=f][k]).
// grid 32 (r), block 256 (f)
__global__ void k_relw(const float* __restrict__ rel, const float* __restrict__ W2,
                       const float* __restrict__ b2, _Float16* __restrict__ H2t) {
  const int r = blockIdx.x, f = threadIdx.x;
  float acc = b2[f];
#pragma unroll 8
  for (int e = 0; e < EE; ++e) acc += rel[r * EE + e] * W2[(D + e) * D + f];
  const _Float16 h = (_Float16)acc;
  for (int b = 0; b < BS; ++b)
    H2t[((size_t)b * D + f) * KCAT + OBJ + r] = h;
}

// ---------------------------------------------------------------------------
// GEMM1: per batch, [128x256] @ [256x512] -> cols 0..255 = h1 (f32),
//        cols 256..511 = h2x stored transposed into H2t[b][n][k=m] (f16)
// grid (8 strips of 64 cols, BS), block 256 = 8 waves; wave w -> rows w*16..
// ---------------------------------------------------------------------------
__global__ void k_gemm1(const _Float16* __restrict__ Xh, const _Float16* __restrict__ WcatT,
                        float* __restrict__ H1, _Float16* __restrict__ H2t) {
  const int b = blockIdx.y;
  const int n0 = blockIdx.x * 64;
  const int m0 = (threadIdx.x >> 5) * 16;
  const int lane = threadIdx.x & 31;
  const _Float16* Xb = Xh + (size_t)b * OBJ * D;
  v8f c[4] = {};
  for (int kt = 0; kt < D; kt += 32) {
    v16h a = load_a16(Xb, D, m0, kt);
#pragma unroll
    for (int j = 0; j < 4; ++j) {
      v16h bb = load_b16t(WcatT, D, n0 + j * 16, kt);
      c[j] = wmma_f16(a, bb, c[j]);
    }
  }
  const int nl = lane & 15, mb = (lane >> 4) * 8;
  if (n0 < D) {
    float* out = H1 + (size_t)b * OBJ * D;
#pragma unroll
    for (int j = 0; j < 4; ++j)
#pragma unroll
      for (int v = 0; v < 8; ++v)
        out[(m0 + mb + v) * D + (n0 + j * 16 + nl)] = c[j][v];
  } else {
    _Float16* out = H2t + (size_t)b * D * KCAT;  // [n][k]
    const int nn0 = n0 - D;
#pragma unroll
    for (int j = 0; j < 4; ++j)
#pragma unroll
      for (int v = 0; v < 8; ++v)
        out[(nn0 + j * 16 + nl) * KCAT + (m0 + mb + v)] = (_Float16)c[j][v];
  }
}

// ---------------------------------------------------------------------------
// GEMM2 + fused epilogue:
//   agg = AsumCat[b](128x160) @ H2[b](160x256)   (B given K-contiguous)
//   xnew = tanh(agg + h1 + b1 + xold);  also write f16 copy for next depth
// grid (4 strips of 64 cols, BS), block 256
// ---------------------------------------------------------------------------
__global__ void k_gemm2(const _Float16* __restrict__ Asum, const _Float16* __restrict__ H2t,
                        const float* __restrict__ H1, const float* __restrict__ b1,
                        const float* __restrict__ xold, float* __restrict__ xnew,
                        _Float16* __restrict__ xh) {
  const int b = blockIdx.y;
  const int n0 = blockIdx.x * 64;
  const int m0 = (threadIdx.x >> 5) * 16;
  const int lane = threadIdx.x & 31;
  const _Float16* Ab = Asum + (size_t)b * OBJ * KCAT;
  const _Float16* Bt = H2t + (size_t)b * D * KCAT;
  v8f c[4] = {};
  for (int kt = 0; kt < KCAT; kt += 32) {
    v16h a = load_a16(Ab, KCAT, m0, kt);
#pragma unroll
    for (int j = 0; j < 4; ++j) {
      v16h bb = load_b16t(Bt, KCAT, n0 + j * 16, kt);
      c[j] = wmma_f16(a, bb, c[j]);
    }
  }
  const int nl = lane & 15, mb = (lane >> 4) * 8;
  const size_t base = (size_t)b * OBJ * D;
#pragma unroll
  for (int j = 0; j < 4; ++j)
#pragma unroll
    for (int v = 0; v < 8; ++v) {
      const int m = m0 + mb + v, n = n0 + j * 16 + nl;
      const size_t idx = base + (size_t)m * D + n;
      const float val = tanhf(c[j][v] + H1[idx] + b1[n] + xold[idx]);
      xnew[idx] = val;
      xh[idx] = (_Float16)val;
    }
}

// ---------------------------------------------------------------------------
extern "C" void kernel_launch(void* const* d_in, const int* in_sizes, int n_in,
                              void* d_out, int out_size, void* d_ws, size_t ws_size,
                              hipStream_t stream) {
  const float* x    = (const float*)d_in[0];  // [32,128,256]
  const float* A    = (const float*)d_in[1];  // [32,128,4096]
  const float* rel  = (const float*)d_in[2];  // [32,64]
  const float* W1   = (const float*)d_in[3];  // [3,256,256]
  const float* b1   = (const float*)d_in[4];  // [3,256]
  const float* W2   = (const float*)d_in[5];  // [3,320,256]
  const float* b2   = (const float*)d_in[6];  // [3,256]
  float* out = (float*)d_out;

  const size_t NX = (size_t)BS * OBJ * D;     // 1,048,576
  char* ws = (char*)d_ws;
  size_t off = 0;
  auto carve = [&](size_t bytes) -> void* {
    bytes = (bytes + 255) & ~(size_t)255;
    void* p = ws + off; off += bytes; return p;
  };
  float*    H1   = (float*)   carve(NX * sizeof(float));                    // 4 MB
  _Float16* H2t  = (_Float16*)carve((size_t)BS * D * KCAT * sizeof(_Float16)); // 2.6 MB
  _Float16* Asum = (_Float16*)carve((size_t)BS * OBJ * KCAT * sizeof(_Float16)); // 1.3 MB
  _Float16* WcT  = (_Float16*)carve((size_t)NCAT * D * sizeof(_Float16));   // 0.26 MB
  _Float16* xh   = (_Float16*)carve(NX * sizeof(_Float16));                 // 2 MB
  float*    xA   = (float*)   carve(NX * sizeof(float));                    // 4 MB
  float*    xB   = (float*)   carve(NX * sizeof(float));                    // 4 MB
  (void)ws_size; (void)in_sizes; (void)n_in; (void)out_size;

  // One-time: A reductions and f16 seed of x
  k_prep_asum<<<dim3(OBJ, BS), 128, 0, stream>>>(A, Asum);
  k_cvt_f16<<<(int)((NX + 255) / 256), 256, 0, stream>>>(x, xh, (int)NX);

  const float* xin = x;
  for (int i = 0; i < DEPTH; ++i) {
    const float* W1i = W1 + (size_t)i * D * D;
    const float* W2i = W2 + (size_t)i * (D + EE) * D;
    k_wcvt<<<NCAT, D, 0, stream>>>(W1i, W2i, WcT);
    k_relw<<<RR, D, 0, stream>>>(rel, W2i, b2 + (size_t)i * D, H2t);
    k_gemm1<<<dim3(8, BS), 256, 0, stream>>>(xh, WcT, H1, H2t);
    float* xo = (i == DEPTH - 1) ? out : (i == 0 ? xA : xB);
    k_gemm2<<<dim3(4, BS), 256, 0, stream>>>(Asum, H2t, H1, b1 + (size_t)i * D,
                                             xin, xo, xh);
    xin = xo;
  }
}
